// DotAttention_67963562492218
// MI455X (gfx1250) — matrix operand order repaired and checked by reading, not compiled, not run on hardware
//
#include <hip/hip_runtime.h>
#include <hip/hip_bf16.h>

typedef __bf16 bf16_t;
typedef __attribute__((ext_vector_type(16))) __bf16 v16bf;
typedef __attribute__((ext_vector_type(8)))  __bf16 v8bf;
typedef __attribute__((ext_vector_type(8)))  float  v8f;

#define NEG_BIG (-1.0e30f)
#define SM_SCALE 0.044194173824159216f  /* 1/sqrt(512) */

template <bool REUSE_A>
__device__ __forceinline__ v8f wmma_bf16(v16bf a, v16bf b, v8f c) {
  return __builtin_amdgcn_wmma_f32_16x16x32_bf16(
      /*neg_a=*/false, a, /*neg_b=*/false, b,
      /*c_mod=*/(short)0, c, /*reuse_a=*/REUSE_A, /*reuse_b=*/false);
}

// Async global->LDS 128-bit copy (per-lane). dst must be an LDS pointer;
// generic LDS addresses carry the LDS byte offset in bits [31:0].
__device__ __forceinline__ void async_copy_b128(void* lds_dst, const void* gsrc) {
  unsigned lds = (unsigned)(size_t)lds_dst;
  unsigned long long ga = (unsigned long long)(size_t)gsrc;
  asm volatile("global_load_async_to_lds_b128 %0, %1, off"
               :: "v"(lds), "v"(ga) : "memory");
}
__device__ __forceinline__ void async_wait0() {
  asm volatile("s_wait_asynccnt 0x0" ::: "memory");
}

// A-fragment (16x32, row-major bf16 source, row stride lda elements).
// Lane layout: row = lane&15; K = {hi*8..hi*8+7, 16+hi*8..16+hi*8+7}.
__device__ __forceinline__ v16bf load_a_bf(const bf16_t* base, int lda, int lane) {
  int r = lane & 15, h8 = (lane >> 4) << 3;
  const bf16_t* p = base + r * lda + h8;
  v8bf lo = *(const v8bf*)(p);
  v8bf hi = *(const v8bf*)(p + 16);
  v16bf a;
#pragma unroll
  for (int j = 0; j < 8; ++j) { a[j] = lo[j]; a[j + 8] = hi[j]; }
  return a;
}

// A-fragment built from f32 LDS tile (convert on the fly).
__device__ __forceinline__ v16bf load_a_f32(const float* base, int lda, int lane) {
  int r = lane & 15, h8 = (lane >> 4) << 3;
  const float* p = base + r * lda + h8;
  v16bf a;
#pragma unroll
  for (int j = 0; j < 8; ++j) { a[j] = (bf16_t)p[j]; a[j + 8] = (bf16_t)p[16 + j]; }
  return a;
}

// B-fragment (32x16). Source laid out so that row n (= output col) has
// contiguous K: element(k,n) = base[n*ldb + k].  Lane: n = lane&15,
// K = {hi*8..hi*8+7, 16+hi*8..16+hi*8+7}.
__device__ __forceinline__ v16bf load_b_bf(const bf16_t* base, int ldb, int lane) {
  int n = lane & 15, h8 = (lane >> 4) << 3;
  const bf16_t* p = base + n * ldb + h8;
  v8bf b0 = *(const v8bf*)(p);
  v8bf b1 = *(const v8bf*)(p + 16);
  v16bf b;
#pragma unroll
  for (int j = 0; j < 8; ++j) { b[j] = b0[j]; b[j + 8] = b1[j]; }
  return b;
}

// ---------------- Phase 0: dtype conversions ----------------
__global__ void cvt_bf16_kernel(const float* __restrict__ x, bf16_t* __restrict__ y, int n) {
  int i = blockIdx.x * blockDim.x + threadIdx.x;
  if (i < n) y[i] = (bf16_t)x[i];
}

// memory [16][2048][512] f32  ->  memT [16][512][2048] bf16
__global__ void transpose_cvt_kernel(const float* __restrict__ M, bf16_t* __restrict__ MT) {
  __shared__ float t[32][33];
  int b = blockIdx.z;
  int m0 = blockIdx.x * 32, d0 = blockIdx.y * 32;
  int tx = threadIdx.x, ty = threadIdx.y;  // 32x8
  const float* src = M + ((size_t)b * 2048 + m0) * 512 + d0;
#pragma unroll
  for (int j = ty; j < 32; j += 8) t[j][tx] = src[(size_t)j * 512 + tx];
  __syncthreads();
  bf16_t* dst = MT + ((size_t)b * 512 + d0) * 2048 + m0;
#pragma unroll
  for (int j = ty; j < 32; j += 8) dst[(size_t)j * 2048 + tx] = (bf16_t)t[tx][j];
}

// ---------------- Phase 1: h = relu(X @ W^T), bf16 out ----------------
// X: [32768][512] f32 ; Wb: [512][512] bf16 (row h contiguous over d)
// grid: 2048  block: 256 (8 waves); wave w does 4 n-tiles at w*64..w*64+63
__global__ void proj_relu_kernel(const float* __restrict__ X,
                                 const bf16_t* __restrict__ Wb,
                                 bf16_t* __restrict__ Y) {
  __shared__ bf16_t sA[16 * 512];  // 16 KB
  int tid = threadIdx.x;
  int row0 = blockIdx.x * 16;
  const float* xp = X + (size_t)row0 * 512;
  for (int i = tid; i < 16 * 512; i += 256) sA[i] = (bf16_t)xp[i];
  __syncthreads();

  int wave = tid >> 5, lane = tid & 31;
  int lane16 = lane & 15, h8 = (lane >> 4) << 3;
  int n0 = wave * 64;

  v8f acc[4];
#pragma unroll
  for (int t = 0; t < 4; ++t)
#pragma unroll
    for (int j = 0; j < 8; ++j) acc[t][j] = 0.f;

#pragma unroll 2
  for (int k = 0; k < 512; k += 32) {
    v16bf a = load_a_bf(sA + k, 512, lane);
    {
      v16bf b = load_b_bf(Wb + (size_t)n0 * 512 + k, 512, lane);
      acc[0] = wmma_bf16<false>(a, b, acc[0]);
    }
#pragma unroll
    for (int t = 1; t < 4; ++t) {
      v16bf b = load_b_bf(Wb + (size_t)(n0 + t * 16) * 512 + k, 512, lane);
      acc[t] = wmma_bf16<true>(a, b, acc[t]);
    }
  }
#pragma unroll
  for (int t = 0; t < 4; ++t) {
    bf16_t* yp = Y + (size_t)(row0 + h8) * 512 + n0 + t * 16 + lane16;
#pragma unroll
    for (int v = 0; v < 8; ++v) {
      float u = acc[t][v];
      yp[(size_t)v * 512] = (bf16_t)(u > 0.f ? u : 0.f);
    }
  }
}

// ---------------- Phase 2: flash attention ----------------
// grid: (Li/16=128, B=16)  block: 256 (8 waves).  Out: attended into
// d_out columns [512,1024) of the concat layout [32768][1024] f32.
__global__ void attn_kernel(const bf16_t* __restrict__ hin,
                            const bf16_t* __restrict__ hmem,
                            const bf16_t* __restrict__ memT,
                            const unsigned char* __restrict__ mask,
                            float* __restrict__ out) {
  constexpr int CH = 128;               // m-chunk
  __shared__ bf16_t sQ[16 * 512];       // 16 KB  query A-tile
  __shared__ float  sS[16 * CH];        //  8 KB  scores
  __shared__ bf16_t sP[16 * CH];        //  4 KB  probs (A-fragment layout)
  __shared__ float  pmax[16][16];
  __shared__ float  psum[16][16];
  __shared__ float  row_m[16], row_l[16], row_a[16];

  int tid = threadIdx.x;
  int b = blockIdx.y;
  int q0 = blockIdx.x * 16;

  { // stage queries (bf16, 16 KB) via async global->LDS b128
    const bf16_t* src = hin + ((size_t)b * 2048 + q0) * 512;
    for (int i = tid; i < 16 * 512 / 8; i += 256)
      async_copy_b128(sQ + i * 8, src + i * 8);
    async_wait0();
  }
  if (tid < 16) { row_m[tid] = -3.0e38f; row_l[tid] = 0.f; }
  __syncthreads();

  int wave = tid >> 5, lane = tid & 31;
  int lane16 = lane & 15, h8 = (lane >> 4) << 3;
  int d0 = wave * 64;                   // this wave's 64-col slice of Dmem
  int r = tid >> 4, c8 = (tid & 15) << 3;  // softmax: 16 threads per row

  const bf16_t* hm_b = hmem + (size_t)b * 2048 * 512;
  const bf16_t* mT_b = memT + (size_t)b * 512 * 2048;
  const unsigned char* mk = mask + (size_t)b * 2048;

  v8f acc[4];
#pragma unroll
  for (int t = 0; t < 4; ++t)
#pragma unroll
    for (int j = 0; j < 8; ++j) acc[t][j] = 0.f;

  for (int m0 = 0; m0 < 2048; m0 += CH) {
    // prefetch next chunk's score-B rows (one line per lane, L2-resident)
    if (m0 + CH < 2048)
      __builtin_prefetch(hm_b + (size_t)(m0 + CH + wave * 16 + lane16) * 512, 0, 1);

    // ---- scores: wave computes 16x16 tile at m = m0 + wave*16, K=512 ----
    int mw = m0 + wave * 16;
    v8f s;
#pragma unroll
    for (int j = 0; j < 8; ++j) s[j] = 0.f;
#pragma unroll 4
    for (int k = 0; k < 512; k += 32) {
      v16bf a = load_a_bf(sQ + k, 512, lane);
      v16bf bb = load_b_bf(hm_b + (size_t)mw * 512 + k, 512, lane);
      s = wmma_bf16<false>(a, bb, s);
    }
    { // scale + mask, write C-layout into sS
      int col = wave * 16 + lane16;
      bool on = mk[m0 + wave * 16 + lane16] != 0;
#pragma unroll
      for (int v = 0; v < 8; ++v) {
        float u = on ? s[v] * SM_SCALE : NEG_BIG;
        sS[(v + h8) * CH + col] = u;
      }
    }
    __syncthreads();

    // ---- online softmax over the 128-wide chunk ----
    float mx = -3.0e38f;
#pragma unroll
    for (int j = 0; j < 8; ++j) mx = fmaxf(mx, sS[r * CH + c8 + j]);
    pmax[r][tid & 15] = mx;
    __syncthreads();
    if (tid < 16) {
      float mo = row_m[tid], mn = mo;
#pragma unroll
      for (int j = 0; j < 16; ++j) mn = fmaxf(mn, pmax[tid][j]);
      row_a[tid] = __expf(mo - mn);
      row_m[tid] = mn;
    }
    __syncthreads();
    float rm = row_m[r];
    float sum = 0.f;
#pragma unroll
    for (int j = 0; j < 8; ++j) {
      float p = __expf(sS[r * CH + c8 + j] - rm);
      sum += p;
      sP[r * CH + c8 + j] = (bf16_t)p;
    }
    psum[r][tid & 15] = sum;
    __syncthreads();
    if (tid < 16) {
      float s2 = 0.f;
#pragma unroll
      for (int j = 0; j < 16; ++j) s2 += psum[tid][j];
      row_l[tid] = row_l[tid] * row_a[tid] + s2;
    }

    // ---- rescale + attended: acc(16x64) += P(16x128) @ memT-slice ----
    float al[8];
#pragma unroll
    for (int v = 0; v < 8; ++v) al[v] = row_a[v + h8];
#pragma unroll
    for (int t = 0; t < 4; ++t)
#pragma unroll
      for (int v = 0; v < 8; ++v) acc[t][v] *= al[v];

#pragma unroll
    for (int kk = 0; kk < CH; kk += 32) {
      v16bf a = load_a_bf(sP + kk, CH, lane);
      {
        v16bf bb = load_b_bf(mT_b + (size_t)d0 * 2048 + m0 + kk, 2048, lane);
        acc[0] = wmma_bf16<false>(a, bb, acc[0]);
      }
#pragma unroll
      for (int t = 1; t < 4; ++t) {
        v16bf bb = load_b_bf(mT_b + (size_t)(d0 + t * 16) * 2048 + m0 + kk, 2048, lane);
        acc[t] = wmma_bf16<true>(a, bb, acc[t]);
      }
    }
    __syncthreads();  // protect sS/sP for next chunk
  }

  // ---- finalize: divide by l, store f32 into concat cols 512.. ----
  float inv[8];
#pragma unroll
  for (int v = 0; v < 8; ++v) inv[v] = 1.0f / row_l[v + h8];
  float* op = out + ((size_t)(b * 2048 + q0 + h8)) * 1024 + 512 + d0 + lane16;
#pragma unroll
  for (int t = 0; t < 4; ++t)
#pragma unroll
    for (int v = 0; v < 8; ++v) op[(size_t)v * 1024 + t * 16] = acc[t][v] * inv[v];
}

// ---------------- Phase 3: gated residual (in place) ----------------
// res = [inputs | attended] (staged f32 in LDS); out = res * sigmoid(res @ Wres^T)
// grid: 2048  block: 256 (8 waves); wave w handles cols [w*128, w*128+128)
__global__ void gate_kernel(const float* __restrict__ inp,
                            const bf16_t* __restrict__ Wres,
                            float* __restrict__ out) {
  __shared__ float sR[16 * 1024];  // 64 KB (<= 320 KB/WGP)
  int tid = threadIdx.x;
  int row0 = blockIdx.x * 16;

  { // async stage res: cols [0,512) from inputs, [512,1024) from attended
    const float* a4 = inp + (size_t)row0 * 512;
    for (int i = tid; i < 2048; i += 256) {
      int rr = i >> 7, cc = (i & 127) * 4;  // 128 float4 per 512-col half
      async_copy_b128(sR + rr * 1024 + cc, a4 + rr * 512 + cc);
      async_copy_b128(sR + rr * 1024 + 512 + cc,
                      out + (size_t)(row0 + rr) * 1024 + 512 + cc);
    }
    async_wait0();
  }
  __syncthreads();

  int wave = tid >> 5, lane = tid & 31;
  int lane16 = lane & 15, h8 = (lane >> 4) << 3;

  v8f acc[8];
#pragma unroll
  for (int t = 0; t < 8; ++t)
#pragma unroll
    for (int j = 0; j < 8; ++j) acc[t][j] = 0.f;

  for (int k = 0; k < 1024; k += 32) {
    v16bf a = load_a_f32(sR + k, 1024, lane);  // built once per k-step
    {
      v16bf b = load_b_bf(Wres + (size_t)(wave * 128) * 1024 + k, 1024, lane);
      acc[0] = wmma_bf16<false>(a, b, acc[0]);
    }
#pragma unroll
    for (int nt = 1; nt < 8; ++nt) {
      int n0 = wave * 128 + nt * 16;
      v16bf b = load_b_bf(Wres + (size_t)n0 * 1024 + k, 1024, lane);
      acc[nt] = wmma_bf16<true>(a, b, acc[nt]);
    }
  }

  float* op = out + (size_t)row0 * 1024;
#pragma unroll
  for (int nt = 0; nt < 8; ++nt) {
    int n0 = wave * 128 + nt * 16;
#pragma unroll
    for (int v = 0; v < 8; ++v) {
      int rr = v + h8;
      float g = 1.0f / (1.0f + __expf(-acc[nt][v]));
      float res = sR[rr * 1024 + n0 + lane16];
      op[(size_t)rr * 1024 + n0 + lane16] = res * g;
    }
  }
}

extern "C" void kernel_launch(void* const* d_in, const int* in_sizes, int n_in,
                              void* d_out, int out_size, void* d_ws, size_t ws_size,
                              hipStream_t stream) {
  (void)in_sizes; (void)n_in; (void)out_size; (void)ws_size;
  const float* inputs = (const float*)d_in[0];          // [16][2048][512]
  const float* memory = (const float*)d_in[1];          // [16][2048][512]
  const unsigned char* mask = (const unsigned char*)d_in[2];  // [16][2048] bool
  const float* W_in  = (const float*)d_in[3];           // [512][512]
  const float* W_mem = (const float*)d_in[4];           // [512][512]
  const float* W_res = (const float*)d_in[5];           // [1024][1024]
  float* out = (float*)d_out;                           // [16][2048][1024]

  bf16_t* ws    = (bf16_t*)d_ws;
  bf16_t* h_in  = ws;                    // 16,777,216 elems (32 MB)
  bf16_t* h_mem = h_in  + 16777216;      // 32 MB
  bf16_t* memT  = h_mem + 16777216;      // 32 MB
  bf16_t* wb_in  = memT  + 16777216;     // 512 KB
  bf16_t* wb_mem = wb_in + 262144;       // 512 KB
  bf16_t* wb_res = wb_mem + 262144;      // 2 MB

  cvt_bf16_kernel<<<1024, 256, 0, stream>>>(W_in,  wb_in,  262144);
  cvt_bf16_kernel<<<1024, 256, 0, stream>>>(W_mem, wb_mem, 262144);
  cvt_bf16_kernel<<<4096, 256, 0, stream>>>(W_res, wb_res, 1048576);
  transpose_cvt_kernel<<<dim3(64, 16, 16), dim3(32, 8), 0, stream>>>(memory, memT);

  proj_relu_kernel<<<2048, 256, 0, stream>>>(inputs, wb_in,  h_in);
  proj_relu_kernel<<<2048, 256, 0, stream>>>(memory, wb_mem, h_mem);

  attn_kernel<<<dim3(128, 16), 256, 0, stream>>>(h_in, h_mem, memT, mask, out);
  gate_kernel<<<2048, 256, 0, stream>>>(inputs, wb_res, out);
}
